// EuclideanCodebook_84877143703693
// MI455X (gfx1250) — matrix operand order repaired and checked by reading, not compiled, not run on hardware
//
#include <hip/hip_runtime.h>
#include <hip/hip_bf16.h>

// ---------------------------------------------------------------------------
// EuclideanCodebook eval forward for MI455X (gfx1250, wave32, WMMA).
//   dist = |x|^2 - 2 x.e^T + |e|^2 ; argmin over 1024 codes; gather + residual.
// GEMM x.e^T via v_wmma_f32_16x16x32_bf16; |x|^2 dropped (row-constant).
// Key MI455X tuning:
//   * LDS codebook rows padded to 528 B -> conflict-free ds_load_b128
//     (each of 64 banks hit exactly 2x per 32-lane b128 = minimum).
//   * 2 A-tiles (32 x rows) per wave kept in VGPRs so each B fragment from
//     LDS feeds 2 WMMAs -> LDS traffic below the 23.3 TB/s HBM floor.
// ---------------------------------------------------------------------------

typedef __attribute__((ext_vector_type(16))) __bf16 v16bf;
typedef __attribute__((ext_vector_type(8)))  float  v8f;

#define DIM      256             // feature dim (GEMM K)
#define CBK      1024            // codebook size (GEMM N)
#define CHUNK    64              // codebook rows staged in LDS per iteration
#define NCHUNKS  (CBK / CHUNK)   // 16
#define GROUPS   (CHUNK / 16)    // 4 column-groups of 16 per chunk
#define KSTEPS   (DIM / 32)      // 8 wmma k-steps per 16x16 output tile
#define LDSROW   (DIM + 8)       // padded LDS row: 264 bf16 = 528 B = 33 uint4
#define ROWVEC   (DIM / 8)       // 32 uint4 of payload per row

union V16LD { uint4 u[2]; v16bf v; };

// ---------------------------------------------------------------------------
// Prep: codebook f32 -> bf16 (row-major, unpadded) + per-row squared norms.
// One block (256 threads) per codebook row.
// ---------------------------------------------------------------------------
__global__ __launch_bounds__(256)
void vq_prep_kernel(const float* __restrict__ embed,
                    __bf16* __restrict__ ebf,
                    float* __restrict__ e2)
{
    const int row = blockIdx.x;
    const int t   = threadIdx.x;
    const float v = embed[(size_t)row * DIM + t];
    ebf[(size_t)row * DIM + t] = (__bf16)v;

    float p = v * v;
    #pragma unroll
    for (int off = 16; off > 0; off >>= 1)
        p += __shfl_down(p, off, 32);

    __shared__ float partial[8];
    const int wave = t >> 5, lane = t & 31;
    if (lane == 0) partial[wave] = p;
    __syncthreads();
    if (t == 0) {
        float s = 0.f;
        #pragma unroll
        for (int i = 0; i < 8; ++i) s += partial[i];
        e2[row] = s;
    }
}

// ---------------------------------------------------------------------------
// Main kernel: 256 threads = 8 waves; each wave owns 2 tiles of 16 x-rows
// (32 rows), so a workgroup covers 256 rows. Grid: nrows/256 blocks (exact
// fit -> uniform control flow, EXEC all ones as WMMA requires).
// ---------------------------------------------------------------------------
__global__ __launch_bounds__(256)
void vq_argmin_kernel(const float* __restrict__ x,
                      const float* __restrict__ embed,   // original f32 codebook
                      const __bf16* __restrict__ ebf,    // bf16 codebook (ws)
                      const float* __restrict__ e2,      // |e|^2 (ws)
                      float* __restrict__ qout,          // (N,256)
                      float* __restrict__ indout,        // (N,)
                      float* __restrict__ resout)        // (N,256)
{
    __shared__ __bf16 sB[CHUNK * LDSROW];                // 33 KB padded chunk

    const int tid  = threadIdx.x;
    const int wave = tid >> 5;
    const int lane = tid & 31;
    const int half = lane >> 4;      // 0: lanes 0-15, 1: lanes 16-31
    const int l15  = lane & 15;
    const int row0 = blockIdx.x * 256 + wave * 32;       // wave's first x row

    // ---- Load two 16x256 A-tiles once, packed per the ISA A layout:
    //      lane(=M) low-half holds K = kb..kb+7 and kb+16..kb+23 (kb = kk*32),
    //      high-half holds K = kb+8..kb+15 and kb+24..kb+31.
    v16bf A0[KSTEPS], A1[KSTEPS];
    {
        const float* xr0 = x + (size_t)(row0 + l15) * DIM;
        const float* xr1 = x + (size_t)(row0 + 16 + l15) * DIM;
        #pragma unroll
        for (int kk = 0; kk < KSTEPS; ++kk) {
            const int kb = kk * 32 + half * 8;
            #pragma unroll
            for (int t = 0; t < 2; ++t) {
                const float* xr = t ? xr1 : xr0;
                const float4 f0 = ((const float4*)(xr + kb))[0];
                const float4 f1 = ((const float4*)(xr + kb))[1];
                const float4 f2 = ((const float4*)(xr + kb + 16))[0];
                const float4 f3 = ((const float4*)(xr + kb + 16))[1];
                v16bf a;
                a[0]=(__bf16)f0.x; a[1]=(__bf16)f0.y; a[2]=(__bf16)f0.z; a[3]=(__bf16)f0.w;
                a[4]=(__bf16)f1.x; a[5]=(__bf16)f1.y; a[6]=(__bf16)f1.z; a[7]=(__bf16)f1.w;
                a[8]=(__bf16)f2.x; a[9]=(__bf16)f2.y; a[10]=(__bf16)f2.z; a[11]=(__bf16)f2.w;
                a[12]=(__bf16)f3.x; a[13]=(__bf16)f3.y; a[14]=(__bf16)f3.z; a[15]=(__bf16)f3.w;
                if (t) A1[kk] = a; else A0[kk] = a;
            }
        }
    }

    // Running (score,index) per lane. In the 16x16 f32 C layout, C VGPR r on a
    // low-half lane is row M=r, high-half row M=r+8; each lane owns the fixed
    // column n = group_base + (lane&15).
    float best0[8], best1[8];
    int   bidx0[8], bidx1[8];
    #pragma unroll
    for (int r = 0; r < 8; ++r) {
        best0[r] = 3.4e38f; bidx0[r] = 0;
        best1[r] = 3.4e38f; bidx1[r] = 0;
    }

    for (int ch = 0; ch < NCHUNKS; ++ch) {
        __syncthreads();                                 // protect previous chunk
        // Stage 64 rows (32 KB payload) into the padded LDS image:
        // 2048 uint4 items, 8 per thread; 32 payload uint4 per 33-uint4 row.
        {
            const uint4* src = (const uint4*)(ebf + (size_t)ch * CHUNK * DIM);
            uint4*       dst = (uint4*)sB;
            #pragma unroll
            for (int i = 0; i < 8; ++i) {
                const int idx = tid + i * 256;
                const int r   = idx >> 5;                // row in chunk
                const int c   = idx & 31;                // uint4 within row
                dst[r * (LDSROW / 8) + c] = src[idx];    // LDSROW/8 = 33
            }
        }
        __syncthreads();

        #pragma unroll
        for (int g = 0; g < GROUPS; ++g) {
            const int nloc = g * 16 + l15;               // lane's column in chunk
            const int n    = ch * CHUNK + nloc;          // global codebook index
            const float en = e2[n];

            // B layout: lane(=N) low-half holds K = kb..kb+7 and kb+16..kb+23,
            // high-half K = kb+8..kb+15 and kb+24..kb+31. Padded row
            // stride (132 dwords) makes both b128s conflict-free.
            v8f c0 = {}, c1 = {};
            const __bf16* bp = sB + nloc * LDSROW;
            #pragma unroll
            for (int kk = 0; kk < KSTEPS; ++kk) {
                V16LD b;
                const uint4* q = (const uint4*)(bp + kk * 32 + half * 8);
                b.u[0] = q[0];
                b.u[1] = q[2];
                c0 = __builtin_amdgcn_wmma_f32_16x16x32_bf16(
                        false, A0[kk], false, b.v, (short)0, c0, false, false);
                c1 = __builtin_amdgcn_wmma_f32_16x16x32_bf16(
                        false, A1[kk], false, b.v, (short)0, c1, false, false);
            }
            #pragma unroll
            for (int r = 0; r < 8; ++r) {
                const float s0 = en - 2.0f * c0[r];      // |e|^2 - 2 x.e
                const bool t0 = s0 < best0[r];
                best0[r] = t0 ? s0 : best0[r];
                bidx0[r] = t0 ? n  : bidx0[r];
                const float s1 = en - 2.0f * c1[r];
                const bool t1 = s1 < best1[r];
                best1[r] = t1 ? s1 : best1[r];
                bidx1[r] = t1 ? n  : bidx1[r];
            }
        }
    }

    // ---- Cross-lane argmin inside each 16-lane half (each half covers all
    // 1024 columns for its 8 rows). Ties -> lower index (jnp.argmin semantics).
    #pragma unroll
    for (int r = 0; r < 8; ++r) {
        #pragma unroll
        for (int off = 1; off < 16; off <<= 1) {
            {
                const float os = __shfl_xor(best0[r], off, 32);
                const int   oi = __shfl_xor(bidx0[r], off, 32);
                const bool take = (os < best0[r]) || (os == best0[r] && oi < bidx0[r]);
                best0[r] = take ? os : best0[r];
                bidx0[r] = take ? oi : bidx0[r];
            }
            {
                const float os = __shfl_xor(best1[r], off, 32);
                const int   oi = __shfl_xor(bidx1[r], off, 32);
                const bool take = (os < best1[r]) || (os == best1[r] && oi < bidx1[r]);
                best1[r] = take ? os : best1[r];
                bidx1[r] = take ? oi : bidx1[r];
            }
        }
    }

    // ---- Emit outputs: per row, 32 lanes cooperatively copy 256 floats
    // (8 per lane as 2x float4) of quantize and residual; lane 0 writes index.
    for (int m = 0; m < 32; ++m) {
        const int tile = m >> 4;                         // which A-tile
        const int r8   = m & 7;
        const int src  = ((m & 15) < 8) ? 0 : 16;        // half holding this row
        const int idx  = tile ? __shfl(bidx1[r8], src, 32)
                              : __shfl(bidx0[r8], src, 32);
        const size_t row = (size_t)row0 + m;
        if (lane == 0) indout[row] = (float)idx;         // int32 argmin value in
                                                         // the float32 out buffer
        const float4* q4 = (const float4*)(embed + (size_t)idx * DIM) + lane * 2;
        const float4* x4 = (const float4*)(x + row * DIM) + lane * 2;
        const float4 q0 = q4[0], q1 = q4[1];
        const float4 v0 = x4[0], v1 = x4[1];
        float4 r0, r1;
        r0.x = q0.x - v0.x; r0.y = q0.y - v0.y; r0.z = q0.z - v0.z; r0.w = q0.w - v0.w;
        r1.x = q1.x - v1.x; r1.y = q1.y - v1.y; r1.z = q1.z - v1.z; r1.w = q1.w - v1.w;
        float4* qo = (float4*)(qout + row * DIM) + lane * 2;
        float4* ro = (float4*)(resout + row * DIM) + lane * 2;
        qo[0] = q0; qo[1] = q1;
        ro[0] = r0; ro[1] = r1;
    }
}

// ---------------------------------------------------------------------------
// Launcher. d_in[0]=x (N*256 f32), d_in[1]=embed (1024*256 f32).
// d_out = [quantize N*256 | embed_ind N | residual N*256] (f32 view).
// d_ws  = [bf16 codebook 512KB | e2 4KB].
// ---------------------------------------------------------------------------
extern "C" void kernel_launch(void* const* d_in, const int* in_sizes, int n_in,
                              void* d_out, int out_size, void* d_ws, size_t ws_size,
                              hipStream_t stream)
{
    const float* x     = (const float*)d_in[0];
    const float* embed = (const float*)d_in[1];
    const int nrows    = in_sizes[0] / DIM;              // 65536

    __bf16* ebf = (__bf16*)d_ws;
    float*  e2  = (float*)((char*)d_ws + (size_t)CBK * DIM * sizeof(__bf16));

    float* qout   = (float*)d_out;
    float* indout = qout + (size_t)nrows * DIM;
    float* resout = indout + nrows;

    vq_prep_kernel<<<CBK, 256, 0, stream>>>(embed, ebf, e2);
    vq_argmin_kernel<<<nrows / 256, 256, 0, stream>>>(
        x, embed, ebf, e2, qout, indout, resout);
}
